// MultiHeadAttention_19842748907896
// MI455X (gfx1250) — matrix operand order repaired and checked by reading, not compiled, not run on hardware
//
#include <hip/hip_runtime.h>

// ---------------- problem constants ----------------
constexpr int cB = 4, cT = 2048, cD = 1024, cH = 16, cHD = 64;

#define DEV __device__ __forceinline__

typedef __attribute__((ext_vector_type(16))) __bf16 v16bf;
typedef __attribute__((ext_vector_type(8)))  float  v8f;
typedef __attribute__((ext_vector_type(4)))  int    v4i;

union FragAB { v16bf v; unsigned u[8]; };
union Acc    { v8f v;   float f[8]; };

// ---------------- bf16 conversion ----------------
DEV unsigned short f2bf(float x) {  // RNE fallback
  unsigned u = __float_as_uint(x);
  u = (u + 0x7FFFu + ((u >> 16) & 1u)) >> 16;
  return (unsigned short)u;
}

#if defined(__has_builtin) && __has_builtin(__builtin_amdgcn_cvt_pk_bf16_f32)
DEV unsigned pk_bf16(float lo, float hi) {  // v_cvt_pk_bf16_f32
  typedef __attribute__((ext_vector_type(2))) __bf16 v2bf;
  union { v2bf v; unsigned u; } c;
  c.v = __builtin_amdgcn_cvt_pk_bf16_f32(lo, hi);
  return c.u;
}
#else
DEV unsigned pk_bf16(float lo, float hi) {
  return (unsigned)f2bf(lo) | ((unsigned)f2bf(hi) << 16);
}
#endif

// ---------------- async global->LDS staging ----------------
#if defined(__has_builtin) && __has_builtin(__builtin_amdgcn_global_load_async_to_lds_b128)
#define HAVE_ASYNC_LDS 1
#endif

DEV void async_copy16(const unsigned short* gsrc, unsigned short* ldst) {
#ifdef HAVE_ASYNC_LDS
  __builtin_amdgcn_global_load_async_to_lds_b128(
      (__attribute__((address_space(1))) v4i*)gsrc,
      (__attribute__((address_space(3))) v4i*)ldst, 0, 0);
#else
  *(uint4*)ldst = *(const uint4*)gsrc;
#endif
}

DEV void async_wait() {
#ifdef HAVE_ASYNC_LDS
#if __has_builtin(__builtin_amdgcn_s_wait_asynccnt)
  __builtin_amdgcn_s_wait_asynccnt(0);
#else
  asm volatile("s_wait_asynccnt 0x0" ::: "memory");
#endif
#endif
}

// -------- WMMA fragment loaders (layouts per CDNA5 ISA 7.12.2) --------
// A (16x32 bf16): lane<16: row=lane, K pairs {0..7,16..23}; lane>=16: K {8..15,24..31}
DEV void frag_a(FragAB& f, const unsigned short* As, int stride, int row) {
  const int lane = threadIdx.x & 31, half = lane >> 4, lr = lane & 15;
  const unsigned short* p = As + (size_t)(row + lr) * stride + 8 * half;
#pragma unroll
  for (int i = 0; i < 4; ++i) {
    f.u[i]     = *(const unsigned*)(p + 2 * i);
    f.u[i + 4] = *(const unsigned*)(p + 16 + 2 * i);
  }
}

// B (32x16 bf16) gathered from transposed LDS tile Bt[n][k]:
// lanes 0-15 hold K {0..7,16..23} (2/dword), lanes 16-31 hold K {8..15,24..31}; N = lane&15
DEV void frag_bt(FragAB& f, const unsigned short* Bt, int stride, int col) {
  const int lane = threadIdx.x & 31, half = lane >> 4, lr = lane & 15;
  const unsigned short* p = Bt + (size_t)(col + lr) * stride + 8 * half;
#pragma unroll
  for (int i = 0; i < 4; ++i) {
    f.u[i]     = *(const unsigned*)(p + 2 * i);
    f.u[i + 4] = *(const unsigned*)(p + 16 + 2 * i);
  }
}

DEV v8f wmma_bf16(const FragAB& a, const FragAB& b, v8f c) {
  return __builtin_amdgcn_wmma_f32_16x16x32_bf16(false, a.v, false, b.v,
                                                 (short)0, c, false, false);
}

// ---------------- fp32 -> bf16 converter (2-wide packed) ----------------
__global__ void f2bf_kernel(const float2* __restrict__ in,
                            unsigned* __restrict__ out, int n2) {
  int i = blockIdx.x * blockDim.x + threadIdx.x;
  if (i < n2) {
    float2 v = in[i];
    out[i] = pk_bf16(v.x, v.y);
  }
}

// ---------------- 128x64 bf16 GEMM tile engine (double-buffered) ----------------
// C[m0:+128, n0:+64] = A[M,K](lda) * B[K,N](ldb); 256 threads = 8 waves,
// wave w owns a 16-row strip; BK=32 per stage; async A staging, ping-pong LDS.
template <bool OUT_F32>
DEV void gemm128x64(const unsigned short* __restrict__ A, int lda,
                    const unsigned short* __restrict__ B, int ldb,
                    void* __restrict__ C, int ldc, const float* __restrict__ bias,
                    int m0, int n0, int K) {
  __shared__ __attribute__((aligned(16))) unsigned short As[2][128 * 32];
  __shared__ __attribute__((aligned(16))) unsigned short Bt[2][64 * 32];

  const int tid = threadIdx.x;
  const int w = tid >> 5, lane = tid & 31, half = lane >> 4, lr = lane & 15;

  auto stage = [&](int buf, int k0) {
    // A tile 128x32: 512 16B chunks, async engine (bypasses VGPRs)
#pragma unroll
    for (int i = 0; i < 2; ++i) {
      int idx = tid + i * 256;               // 0..511
      int r = idx >> 2, j = (idx & 3) << 3;  // j = 0,8,16,24
      async_copy16(A + (size_t)(m0 + r) * lda + k0 + j, &As[buf][r * 32 + j]);
    }
    // B tile 32x64, transposed into Bt[n][k] (needs VGPR pass)
#pragma unroll
    for (int i = 0; i < 4; ++i) {
      int idx = tid + i * 256;               // 0..1023
      int kk = idx >> 5, nc = (idx & 31) << 1;
      unsigned d = *(const unsigned*)(B + (size_t)(k0 + kk) * ldb + n0 + nc);
      Bt[buf][nc * 32 + kk]       = (unsigned short)(d & 0xFFFFu);
      Bt[buf][(nc + 1) * 32 + kk] = (unsigned short)(d >> 16);
    }
  };

  Acc acc[4];
#pragma unroll
  for (int t = 0; t < 4; ++t)
#pragma unroll
    for (int g = 0; g < 8; ++g) acc[t].f[g] = 0.0f;

  stage(0, 0);
  async_wait();
  __syncthreads();

  for (int k0 = 0; k0 < K; k0 += 32) {
    const int cur = (k0 >> 5) & 1;
    if (k0 + 32 < K) stage(cur ^ 1, k0 + 32);  // prefetch next stage

    FragAB a; frag_a(a, As[cur], 32, w * 16);
#pragma unroll
    for (int t = 0; t < 4; ++t) {
      FragAB b; frag_bt(b, Bt[cur], 32, t * 16);
      acc[t].v = wmma_bf16(a, b, acc[t].v);
    }
    async_wait();
    __syncthreads();
  }

  // epilogue: C layout -> row = g + 8*half, col = lane&15 (per 16-wide tile)
#pragma unroll
  for (int t = 0; t < 4; ++t) {
    int col = n0 + t * 16 + lr;
#pragma unroll
    for (int g = 0; g < 8; g += 2) {
      int row = m0 + w * 16 + g + 8 * half;
      if (OUT_F32) {
        ((float*)C)[(size_t)row * ldc + col]       = acc[t].f[g] + bias[col];
        ((float*)C)[(size_t)(row + 1) * ldc + col] = acc[t].f[g + 1] + bias[col];
      } else {
        unsigned pk = pk_bf16(acc[t].f[g], acc[t].f[g + 1]);
        ((unsigned short*)C)[(size_t)row * ldc + col]       = (unsigned short)pk;
        ((unsigned short*)C)[(size_t)(row + 1) * ldc + col] = (unsigned short)(pk >> 16);
      }
    }
  }
}

// ---------------- kernel 1: fused QKV projections ----------------
// grid: (T/128, H, 3*B); z = b*3 + proj
__global__ __launch_bounds__(256) void qkv_gemm_kernel(
    const unsigned short* __restrict__ xb, const unsigned short* __restrict__ wqb,
    const unsigned short* __restrict__ wkb, const unsigned short* __restrict__ wvb,
    unsigned short* __restrict__ q, unsigned short* __restrict__ k,
    unsigned short* __restrict__ v) {
  const int b = blockIdx.z / 3, p = blockIdx.z % 3, h = blockIdx.y;
  const int m0 = blockIdx.x * 128;
  const unsigned short* W = (p == 0 ? wqb : p == 1 ? wkb : wvb) +
                            (size_t)h * cD * cHD;
  unsigned short* O = (p == 0 ? q : p == 1 ? k : v) +
                      ((size_t)b * cH + h) * cT * cHD;
  const unsigned short* A = xb + (size_t)b * cT * cD;
  gemm128x64<false>(A, cD, W, cHD, O, cHD, nullptr, m0, 0, cD);
}

// ---------------- kernel 2: flash attention ----------------
// grid: (T/64, H, B), 128 threads = 4 waves; wave w owns q rows [qb0+16w, +16)
__global__ __launch_bounds__(128) void flash_attn_kernel(
    const unsigned short* __restrict__ q, const unsigned short* __restrict__ k,
    const unsigned short* __restrict__ v, unsigned short* __restrict__ att) {
  constexpr float SCALE = 0.125f;  // 1/sqrt(64)
  __shared__ __attribute__((aligned(16))) unsigned short Qs[64 * 64];
  __shared__ __attribute__((aligned(16))) unsigned short Ks[32 * 64];
  __shared__ __attribute__((aligned(16))) unsigned short Vt[64 * 32];
  __shared__ __attribute__((aligned(16))) unsigned short Ps[4][16 * 32];

  const int tid = threadIdx.x;
  const int w = tid >> 5, lane = tid & 31, half = lane >> 4, lr = lane & 15;
  const int b = blockIdx.z, h = blockIdx.y, qb0 = blockIdx.x * 64;
  const size_t bh = ((size_t)b * cH + h) * (size_t)cT * cHD;

  // stage Q block (64x64 bf16) once via async engine: 512 chunks / 128 thr
#pragma unroll
  for (int i = 0; i < 4; ++i) {
    int idx = tid + i * 128;               // 0..511
    int r = idx >> 3, j = (idx & 7) << 3;  // 8 chunks per 64-elem row
    async_copy16(q + bh + (size_t)(qb0 + r) * cHD + j, &Qs[r * 64 + j]);
  }

  float m[8], l[8];
  Acc o[4];
#pragma unroll
  for (int g = 0; g < 8; ++g) { m[g] = -1e30f; l[g] = 0.0f; }
#pragma unroll
  for (int t = 0; t < 4; ++t)
#pragma unroll
    for (int g = 0; g < 8; ++g) o[t].f[g] = 0.0f;

  for (int kv0 = 0; kv0 < qb0 + 64; kv0 += 32) {  // causal loop bound
    __syncthreads();  // safe to overwrite Ks/Vt
    // K block: natural [kv][hd] layout == transposed-B layout for Q*K^T; async
#pragma unroll
    for (int i = 0; i < 2; ++i) {
      int idx = tid + i * 128;             // 0..255
      int r = idx >> 3, j = (idx & 7) << 3;
      async_copy16(k + bh + (size_t)(kv0 + r) * cHD + j, &Ks[r * 64 + j]);
    }
    // V block transposed into Vt[hd][kv] (VGPR pass for the transpose)
#pragma unroll
    for (int i = 0; i < 8; ++i) {
      int idx = tid + i * 128;             // 0..1023
      int r = idx >> 5, c = (idx & 31) << 1;
      unsigned d = *(const unsigned*)(v + bh + (size_t)(kv0 + r) * cHD + c);
      Vt[c * 32 + r]       = (unsigned short)(d & 0xFFFFu);
      Vt[(c + 1) * 32 + r] = (unsigned short)(d >> 16);
    }
    async_wait();
    __syncthreads();

    // S = Q (16x64) * K^T (64x32): 2 K-steps x 2 N-tiles
    Acc s0, s1;
#pragma unroll
    for (int g = 0; g < 8; ++g) { s0.f[g] = 0.0f; s1.f[g] = 0.0f; }
#pragma unroll
    for (int ks = 0; ks < 64; ks += 32) {
      FragAB a;  frag_a(a, Qs + ks, 64, w * 16);
      FragAB b0; frag_bt(b0, Ks + ks, 64, 0);
      s0.v = wmma_bf16(a, b0, s0.v);
      FragAB b1; frag_bt(b1, Ks + ks, 64, 16);
      s1.v = wmma_bf16(a, b1, s1.v);
    }

    // online softmax; row r = g + 8*half lives in one 16-lane half
    const int qrb = qb0 + w * 16 + 8 * half;
    const int c0 = kv0 + lr, c1 = kv0 + 16 + lr;
#pragma unroll
    for (int g = 0; g < 8; ++g) {
      int qr = qrb + g;
      float f0 = (c0 <= qr) ? s0.f[g] * SCALE : -1e30f;
      float f1 = (c1 <= qr) ? s1.f[g] * SCALE : -1e30f;
      float rm = fmaxf(f0, f1);
#pragma unroll
      for (int d = 8; d >= 1; d >>= 1) rm = fmaxf(rm, __shfl_xor(rm, d, 32));
      float mn = fmaxf(m[g], rm);
      float al = __expf(m[g] - mn);
      float p0 = __expf(f0 - mn);
      float p1 = __expf(f1 - mn);
      float rs = p0 + p1;
#pragma unroll
      for (int d = 8; d >= 1; d >>= 1) rs += __shfl_xor(rs, d, 32);
      l[g] = l[g] * al + rs;
      m[g] = mn;
#pragma unroll
      for (int t = 0; t < 4; ++t) o[t].f[g] *= al;
      // C-layout -> A-layout transpose through per-wave LDS tile (bf16)
      unsigned pk = pk_bf16(p0, p1);
      Ps[w][(g + 8 * half) * 32 + lr]      = (unsigned short)pk;
      Ps[w][(g + 8 * half) * 32 + 16 + lr] = (unsigned short)(pk >> 16);
    }
    // wave-local (LDS in-order within wave): O += P(16x32) * V(32x64)
    FragAB ap; frag_a(ap, Ps[w], 32, 0);
#pragma unroll
    for (int t = 0; t < 4; ++t) {
      FragAB bv; frag_bt(bv, Vt, 32, t * 16);
      o[t].v = wmma_bf16(ap, bv, o[t].v);
    }
  }

  // normalize and write att[b][t][h*64 + e] (bf16, heads concatenated)
#pragma unroll
  for (int g = 0; g < 8; g += 2) {
    float i0 = 1.0f / l[g], i1 = 1.0f / l[g + 1];
    int row = qb0 + w * 16 + g + 8 * half;
    size_t base0 = ((size_t)b * cT + row) * (size_t)(cH * cHD) + (size_t)h * cHD;
    size_t base1 = base0 + (size_t)(cH * cHD);
#pragma unroll
    for (int t = 0; t < 4; ++t) {
      unsigned pk = pk_bf16(o[t].f[g] * i0, o[t].f[g + 1] * i1);
      att[base0 + t * 16 + lr] = (unsigned short)pk;
      att[base1 + t * 16 + lr] = (unsigned short)(pk >> 16);
    }
  }
}

// ---------------- kernel 3: output projection ----------------
// grid: (B*T/128, D/64)
__global__ __launch_bounds__(256) void proj_gemm_kernel(
    const unsigned short* __restrict__ att, const unsigned short* __restrict__ wpb,
    const float* __restrict__ bp, float* __restrict__ out) {
  gemm128x64<true>(att, cD, wpb, cD, out, cD, bp,
                   blockIdx.x * 128, blockIdx.y * 64, cD);
}

// ---------------- host launcher ----------------
extern "C" void kernel_launch(void* const* d_in, const int* in_sizes, int n_in,
                              void* d_out, int out_size, void* d_ws, size_t ws_size,
                              hipStream_t stream) {
  const float* x  = (const float*)d_in[0];
  const float* Wq = (const float*)d_in[1];
  const float* Wk = (const float*)d_in[2];
  const float* Wv = (const float*)d_in[3];
  const float* Wp = (const float*)d_in[4];
  const float* bp = (const float*)d_in[5];
  float* out = (float*)d_out;

  // workspace layout (bf16 halves intermediate HBM traffic); ~92.3 MB total
  char* ws = (char*)d_ws;
  const size_t nX   = (size_t)cB * cT * cD;
  const size_t nW   = (size_t)cH * cD * cHD;
  const size_t nWp  = (size_t)cD * cD;
  const size_t nQKV = (size_t)cB * cH * cT * cHD;
  unsigned short* xb   = (unsigned short*)ws;  ws += nX   * 2;
  unsigned short* wqb  = (unsigned short*)ws;  ws += nW   * 2;
  unsigned short* wkb  = (unsigned short*)ws;  ws += nW   * 2;
  unsigned short* wvb  = (unsigned short*)ws;  ws += nW   * 2;
  unsigned short* wpb  = (unsigned short*)ws;  ws += nWp  * 2;
  unsigned short* qb   = (unsigned short*)ws;  ws += nQKV * 2;
  unsigned short* kb   = (unsigned short*)ws;  ws += nQKV * 2;
  unsigned short* vb   = (unsigned short*)ws;  ws += nQKV * 2;
  unsigned short* attb = (unsigned short*)ws;  ws += nX   * 2;

  auto conv = [&](const float* src, unsigned short* dst, size_t n) {
    int n2 = (int)(n / 2);
    f2bf_kernel<<<dim3((unsigned)((n2 + 255) / 256)), 256, 0, stream>>>(
        (const float2*)src, (unsigned*)dst, n2);
  };
  conv(x, xb, nX);
  conv(Wq, wqb, nW);
  conv(Wk, wkb, nW);
  conv(Wv, wvb, nW);
  conv(Wp, wpb, nWp);

  dim3 g1(cT / 128, cH, 3 * cB);
  qkv_gemm_kernel<<<g1, 256, 0, stream>>>(xb, wqb, wkb, wvb, qb, kb, vb);

  dim3 g2(cT / 64, cH, cB);
  flash_attn_kernel<<<g2, 128, 0, stream>>>(qb, kb, vb, attb);

  dim3 g3((cB * cT) / 128, cD / 64);
  proj_gemm_kernel<<<g3, 256, 0, stream>>>(attb, wpb, bp, out);
}
